// SelfAttentionLayer_36575941493295
// MI455X (gfx1250) — matrix operand order repaired and checked by reading, not compiled, not run on hardware
//
#include <hip/hip_runtime.h>
#include <hip/hip_bf16.h>

// ---------------------------------------------------------------------------
// SelfAttention: scores = X X^T (per batch), softmax rows, out = x * attn
// B=16, N=D=1024, fp32 in/out.
// Split-bf16 (hi+lo) WMMA GEMM for ~fp32-accurate scores at bf16 matrix rate.
// ---------------------------------------------------------------------------

#define NDIM 1024
#define NN   (1024 * 1024)
#define NBATCH 16

typedef __attribute__((ext_vector_type(16))) __bf16 v16bf;
typedef __attribute__((ext_vector_type(8)))  float  v8f;

static constexpr int A_STRIDE = 1032;  // bf16 elems per LDS row (pad: banks)
static constexpr int S_STRIDE = 1028;  // floats per LDS score row (pad: banks)

__device__ __forceinline__ unsigned short f2bf(float f) {
    unsigned u = __float_as_uint(f);
    unsigned r = u + 0x7FFFu + ((u >> 16) & 1u);   // round-to-nearest-even
    return (unsigned short)(r >> 16);
}
__device__ __forceinline__ float bf2f(unsigned short h) {
    return __uint_as_float(((unsigned)h) << 16);
}

union BF16x16 { uint4 u[2]; v16bf v; };

// ---- pass 0: split fp32 -> (hi, lo) bf16 planes ---------------------------
__global__ void split_bf16_kernel(const float* __restrict__ x,
                                  unsigned short* __restrict__ xh,
                                  unsigned short* __restrict__ xl, int n4) {
    int gid = blockIdx.x * blockDim.x + threadIdx.x;
    if (gid >= n4) return;
    float4 f = ((const float4*)x)[gid];
    float fx[4] = {f.x, f.y, f.z, f.w};
    ushort4 h, l;
    unsigned short* hp = &h.x;
    unsigned short* lp = &l.x;
#pragma unroll
    for (int i = 0; i < 4; ++i) {
        unsigned short hb = f2bf(fx[i]);
        hp[i] = hb;
        lp[i] = f2bf(fx[i] - bf2f(hb));
    }
    ((ushort4*)xh)[gid] = h;
    ((ushort4*)xl)[gid] = l;
}

// ---- fused scores + softmax + elementwise multiply ------------------------
// grid: 1024 blocks (batch * 64 row-tiles), 128 threads (4 waves).
// wave w computes column tiles jt = w*16 .. w*16+15 of its 16-row score slab.
__global__ __launch_bounds__(128) void attn_kernel(
        const float* __restrict__ x,
        const unsigned short* __restrict__ xh,
        const unsigned short* __restrict__ xl,
        float* __restrict__ out) {
    extern __shared__ char smem[];
    unsigned short* Ah = (unsigned short*)smem;                 // 16 x A_STRIDE
    unsigned short* Al = Ah + 16 * A_STRIDE;                    // 16 x A_STRIDE
    float* S      = (float*)(Al + 16 * A_STRIDE);               // 16 x S_STRIDE
    float* rowMax = S + 16 * S_STRIDE;                          // 16
    float* rowInv = rowMax + 16;                                // 16

    const int t   = threadIdx.x;
    const int bid = blockIdx.x;
    const int bb  = bid >> 6;
    const int i0  = (bid & 63) << 4;
    const size_t xbase = (size_t)bb * NN;

    // ---- stage A tile (hi+lo) into LDS, cooperative, b128 granularity ----
    {
        const uint4* gh = (const uint4*)(xh + xbase + (size_t)i0 * NDIM);
        const uint4* gl = (const uint4*)(xl + xbase + (size_t)i0 * NDIM);
        for (int q = t; q < 2048; q += 128) {   // 16 rows * 128 uint4/row
            int m = q >> 7;
            int c = q & 127;                    // uint4 within row (8 bf16)
            uint4 vh = gh[q];
            uint4 vl = gl[q];
            *(uint4*)(Ah + m * A_STRIDE + c * 8) = vh;
            *(uint4*)(Al + m * A_STRIDE + c * 8) = vl;
        }
    }
    __syncthreads();

    // ---- pass 1: S[16 x 1024] via split-bf16 WMMA ----
    const int lane = t & 31;
    const int wv   = t >> 5;
    const int ln   = lane & 15;
    const int hih  = lane >> 4;          // half-wave select
    // A register layout: lane = row, elems 0..7 = K (+0 | +8), elems 8..15 = +16
    const uint4* ah0 = (const uint4*)(Ah + ln * A_STRIDE + hih * 8);
    const uint4* al0 = (const uint4*)(Al + ln * A_STRIDE + hih * 8);

    for (int tt = 0; tt < 16; ++tt) {
        const int j0 = (wv * 16 + tt) * 16;
        // B register layout: lane = column (row j0+ln of X), same K order as A:
        // elems 0..7 = K (+0 | +8), elems 8..15 = +16.
        const uint4* bh = (const uint4*)(xh + xbase + (size_t)(j0 + ln) * NDIM + hih * 8);
        const uint4* bl = (const uint4*)(xl + xbase + (size_t)(j0 + ln) * NDIM + hih * 8);

        v8f acc = {};
#pragma unroll 4
        for (int kt = 0; kt < 32; ++kt) {
            BF16x16 Avh, Avl, Bvh, Bvl;
            Avh.u[0] = ah0[kt * 4];     Avh.u[1] = ah0[kt * 4 + 2];
            Avl.u[0] = al0[kt * 4];     Avl.u[1] = al0[kt * 4 + 2];
            Bvh.u[0] = bh[kt * 4];      Bvh.u[1] = bh[kt * 4 + 2];
            Bvl.u[0] = bl[kt * 4];      Bvl.u[1] = bl[kt * 4 + 2];
            acc = __builtin_amdgcn_wmma_f32_16x16x32_bf16(
                      false, Avh.v, false, Bvh.v, (short)0, acc, false, false);
            acc = __builtin_amdgcn_wmma_f32_16x16x32_bf16(
                      false, Avh.v, false, Bvl.v, (short)0, acc, false, false);
            acc = __builtin_amdgcn_wmma_f32_16x16x32_bf16(
                      false, Avl.v, false, Bvh.v, (short)0, acc, false, false);
        }
        // C layout: vgpr r, lane -> row r + 8*(lane>=16), col = lane%16
        const int colS  = j0 + ln;
        const int rbase = hih * 8;
#pragma unroll
        for (int r = 0; r < 8; ++r)
            S[(rbase + r) * S_STRIDE + colS] = acc[r];
    }
    __syncthreads();

    // ---- row max (8 threads per row, stride-8 columns) ----
    {
        const int m = t >> 3, sub = t & 7;
        const float* Sr = S + m * S_STRIDE;
        float pm = -3.0e38f;
        for (int k = 0; k < 128; ++k) pm = fmaxf(pm, Sr[sub + 8 * k]);
        pm = fmaxf(pm, __shfl_xor(pm, 1, 32));
        pm = fmaxf(pm, __shfl_xor(pm, 2, 32));
        pm = fmaxf(pm, __shfl_xor(pm, 4, 32));
        if (sub == 0) rowMax[m] = pm;
    }
    __syncthreads();

    // ---- row sum of exp ----
    {
        const int m = t >> 3, sub = t & 7;
        const float rm = rowMax[m];
        const float* Sr = S + m * S_STRIDE;
        float ps = 0.f;
        for (int k = 0; k < 128; ++k) ps += __expf(Sr[sub + 8 * k] - rm);
        ps += __shfl_xor(ps, 1, 32);
        ps += __shfl_xor(ps, 2, 32);
        ps += __shfl_xor(ps, 4, 32);
        if (sub == 0) rowInv[m] = 1.0f / ps;
    }
    __syncthreads();

    // ---- pass 2: out = x * softmax(S), coalesced float4 ----
    {
        const float4* xg4 = (const float4*)(x + xbase + (size_t)i0 * NDIM);
        float4*       og4 = (float4*)(out + xbase + (size_t)i0 * NDIM);
        for (int k = 0; k < 32; ++k) {
            int q  = k * 128 + t;        // float4 index, 256 per row
            int m  = q >> 8;
            int j4 = (q & 255) * 4;
            float rm = rowMax[m], ri = rowInv[m];
            const float* Sr = S + m * S_STRIDE + j4;
            float4 xv = xg4[q];
            float4 o;
            o.x = xv.x * (__expf(Sr[0] - rm) * ri);
            o.y = xv.y * (__expf(Sr[1] - rm) * ri);
            o.z = xv.z * (__expf(Sr[2] - rm) * ri);
            o.w = xv.w * (__expf(Sr[3] - rm) * ri);
            og4[q] = o;
        }
    }
}

extern "C" void kernel_launch(void* const* d_in, const int* in_sizes, int n_in,
                              void* d_out, int out_size, void* d_ws, size_t ws_size,
                              hipStream_t stream) {
    (void)in_sizes; (void)n_in; (void)out_size; (void)ws_size;
    const float* x   = (const float*)d_in[0];
    float*       out = (float*)d_out;

    // workspace: hi plane (32 MiB) then lo plane (32 MiB)
    unsigned short* xh = (unsigned short*)d_ws;
    unsigned short* xl = xh + (size_t)NBATCH * NN;

    const int n4 = NBATCH * NN / 4;
    split_bf16_kernel<<<(n4 + 255) / 256, 256, 0, stream>>>(x, xh, xl, n4);

    const size_t lds_bytes = (size_t)(2 * 16 * A_STRIDE) * sizeof(unsigned short)
                           + (size_t)(16 * S_STRIDE) * sizeof(float)
                           + 32 * sizeof(float);
    (void)hipFuncSetAttribute((const void*)attn_kernel,
                              hipFuncAttributeMaxDynamicSharedMemorySize,
                              (int)lds_bytes);
    attn_kernel<<<NBATCH * 64, 128, lds_bytes, stream>>>(x, xh, xl, out);
}
